// TemporalUNetMoE_60370060312674
// MI455X (gfx1250) — matrix operand order repaired and checked by reading, not compiled, not run on hardware
//
#include <hip/hip_runtime.h>
#include <hip/hip_bf16.h>
#include <math.h>

// Problem constants (from reference)
#define D_MODEL 512
#define SEQ     512
#define NB      8
#define T_TOK   4096      // NB*SEQ
#define N_HEADS 4
#define HEAD_D  128
#define FF_DIM  2048
#define N_EXP   8
#define TD_DIM  768
#define IN_CH   144       // BONES*CH
#define OUT_ELEMS (T_TOK*IN_CH)

typedef __attribute__((ext_vector_type(16))) __bf16 v16bf;
typedef __attribute__((ext_vector_type(8)))  float  v8f;

union alignas(32) Frag16 { unsigned int u[8]; v16bf v; uint4 q[2]; };

__device__ __forceinline__ unsigned short f2bf(float x) {
  unsigned int u = __float_as_uint(x);
  u += 0x7FFFu + ((u >> 16) & 1u);           // round-to-nearest-even
  return (unsigned short)(u >> 16);
}
__device__ __forceinline__ unsigned int pack2_bf16(float x, float y) {
  return (unsigned int)f2bf(x) | ((unsigned int)f2bf(y) << 16);
}
__device__ __forceinline__ float bf2f(unsigned short h) {
  return __uint_as_float(((unsigned int)h) << 16);
}

// ---------------------------------------------------------------------------
// WMMA GEMM on bf16 operands: C[M,N] (+)= epilogue( alpha * A @ B )
//   A: bf16 row-major [M,K] (lda). B: bf16 row-major [K,N] (ldb).
//   Per wave: 32x64 tile = 2 A-frags x 4 B-frags = 8 WMMAs per K-step,
//   4 x b128 A loads + 64 u16 B loads per K-step. Interior tiles are branch-free.
//   C: f32 (ACCUM adds) or bf16 (OUT16).
// ---------------------------------------------------------------------------
template<int GELU, int ACCUM, int BIAS, int RSCALE, int OUT16>
__global__ __launch_bounds__(128) void gemm_wmma(
    const unsigned short* __restrict__ A, const unsigned short* __restrict__ B,
    void* __restrict__ Cv, const float* __restrict__ bias,
    const float* __restrict__ rowscale, int rs_stride,
    int M, int N, int K, int lda, int ldb, int ldc, float alpha,
    int headsZ, long sAb, long sAh, long sBb, long sBh, long sCb, long sCh)
{
  const int lane = threadIdx.x & 31;
  const int wave = threadIdx.x >> 5;
  const int zb = (int)blockIdx.z / headsZ;
  const int zh = (int)blockIdx.z % headsZ;
  A += (long)zb * sAb + (long)zh * sAh;
  B += (long)zb * sBb + (long)zh * sBh;
  const long coff = (long)zb * sCb + (long)zh * sCh;

  const int m0 = ((int)blockIdx.y * 4 + wave) * 32;
  const int n0 = (int)blockIdx.x * 64;
  if (m0 >= M) return;                       // uniform per wave

  const int mr   = lane & 15;
  const int half = lane >> 4;

  v8f zero = {0.f,0.f,0.f,0.f,0.f,0.f,0.f,0.f};
  v8f acc[2][4];
#pragma unroll
  for (int mi = 0; mi < 2; ++mi)
#pragma unroll
    for (int j = 0; j < 4; ++j) acc[mi][j] = zero;

  const bool interior = (m0 + 32 <= M) && (n0 + 64 <= N);
  const int  Kmain    = interior ? (K & ~31) : 0;
  const long abase0   = (long)(m0 + mr) * lda + half * 8;
  const long abase1   = (long)(m0 + 16 + mr) * lda + half * 8;

  // ---- fast path: fully interior, vectorized 16B loads, no branches ----
  for (int k0 = 0; k0 < Kmain; k0 += 32) {
    Frag16 a0, a1;
    a0.q[0] = *(const uint4*)(A + abase0 + k0);
    a0.q[1] = *(const uint4*)(A + abase0 + k0 + 16);
    a1.q[0] = *(const uint4*)(A + abase1 + k0);
    a1.q[1] = *(const uint4*)(A + abase1 + k0 + 16);
    const long bcol = n0 + mr;   // B: lane carries N, elements carry K
#pragma unroll
    for (int j = 0; j < 4; ++j) {
      Frag16 b;
#pragma unroll
      for (int v = 0; v < 8; ++v) {
        int kb = k0 + half * 8 + ((v & 4) ? 16 + (v & 3) * 2 : v * 2);
        b.u[v] = (unsigned int)B[(long)kb * ldb + bcol + j * 16] |
                 ((unsigned int)B[(long)(kb + 1) * ldb + bcol + j * 16] << 16);
      }
      acc[0][j] = __builtin_amdgcn_wmma_f32_16x16x32_bf16(
                      false, a0.v, false, b.v, (short)0, acc[0][j], false, false);
      acc[1][j] = __builtin_amdgcn_wmma_f32_16x16x32_bf16(
                      false, a1.v, false, b.v, (short)0, acc[1][j], false, false);
    }
  }

  // ---- guarded path: edge tiles and K tails ----
  for (int k0 = Kmain; k0 < K; k0 += 32) {
    Frag16 a[2];
#pragma unroll
    for (int mi = 0; mi < 2; ++mi) {
      const int  mrow = m0 + mi * 16 + mr;
      const bool mOK  = mrow < M;
      const long arow = (long)mrow * lda;
#pragma unroll
      for (int v = 0; v < 8; ++v) {
        int kb = k0 + half * 8 + ((v & 4) ? 16 + (v & 3) * 2 : v * 2);
        unsigned int lo = (mOK && kb < K)     ? A[arow + kb]     : 0u;
        unsigned int hi = (mOK && kb + 1 < K) ? A[arow + kb + 1] : 0u;
        a[mi].u[v] = lo | (hi << 16);
      }
    }
    const int kh = k0 + half * 8;
#pragma unroll
    for (int j = 0; j < 4; ++j) {
      Frag16 b;
#pragma unroll
      for (int v = 0; v < 8; ++v) {
        int kb = kh + ((v & 4) ? 16 + (v & 3) * 2 : v * 2);
        int n = n0 + j * 16 + mr;
        unsigned int lo = 0u, hi = 0u;
        if (n < N) {
          if (kb < K)     lo = B[(long)kb * ldb + n];
          if (kb + 1 < K) hi = B[(long)(kb + 1) * ldb + n];
        }
        b.u[v] = lo | (hi << 16);
      }
      acc[0][j] = __builtin_amdgcn_wmma_f32_16x16x32_bf16(
                      false, a[0].v, false, b.v, (short)0, acc[0][j], false, false);
      acc[1][j] = __builtin_amdgcn_wmma_f32_16x16x32_bf16(
                      false, a[1].v, false, b.v, (short)0, acc[1][j], false, false);
    }
  }

  // ---- epilogue: C layout lanes0-15 M=r, lanes16-31 M=r+8, N=lane&15 ----
  const int nl = lane & 15;
  float* Cf = (float*)Cv;
  unsigned short* Ch = (unsigned short*)Cv;
#pragma unroll
  for (int mi = 0; mi < 2; ++mi) {
#pragma unroll
    for (int j = 0; j < 4; ++j) {
      int ng = n0 + j * 16 + nl;
      if (ng >= N) continue;
#pragma unroll
      for (int r = 0; r < 8; ++r) {
        int mg = m0 + mi * 16 + r + half * 8;
        if (mg >= M) continue;
        float val = acc[mi][j][r] * alpha;
        if (BIAS)   val += bias[ng];
        if (GELU)   val = 0.5f * val * (1.0f + erff(val * 0.70710678118654752f));
        if (RSCALE) val *= rowscale[(long)mg * rs_stride];
        long ci = coff + (long)mg * ldc + ng;
        if (ACCUM)      Cf[ci] += val;
        else if (OUT16) Ch[ci] = f2bf(val);
        else            Cf[ci] = val;
      }
    }
  }
}

// ---------------------------------------------------------------------------
// bf16 512x512 per-batch transpose (LDS tiled): out[b][d][t] = in[b][t][d]
// Turns the attention score GEMM into pure NN form.
// ---------------------------------------------------------------------------
__global__ __launch_bounds__(256) void transpose512_kernel(
    const unsigned short* __restrict__ in, unsigned short* __restrict__ out) {
  __shared__ unsigned short tile[32][33];
  int b = blockIdx.z;
  int tx = threadIdx.x;              // 0..31
  int ty = threadIdx.y;              // 0..7
  int x0 = blockIdx.x * 32, y0 = blockIdx.y * 32;
  const unsigned short* src = in + (long)b * 512 * 512;
  unsigned short* dst = out + (long)b * 512 * 512;
#pragma unroll
  for (int i = 0; i < 32; i += 8)
    tile[ty + i][tx] = src[(long)(y0 + ty + i) * 512 + x0 + tx];
  __syncthreads();
#pragma unroll
  for (int i = 0; i < 32; i += 8)
    dst[(long)(x0 + ty + i) * 512 + y0 + tx] = tile[tx][ty + i];
}

// ---------------------------------------------------------------------------
// Elementwise / reduction kernels
// ---------------------------------------------------------------------------
__global__ void cvt_bf16_kernel(const float* __restrict__ src,
                                unsigned short* __restrict__ dst, int n4) {
  int i = blockIdx.x * blockDim.x + threadIdx.x;
  if (i >= n4) return;
  float4 v = ((const float4*)src)[i];
  ((uint2*)dst)[i] = make_uint2(pack2_bf16(v.x, v.y), pack2_bf16(v.z, v.w));
}

__global__ void ewadd_bf16_kernel(const float* __restrict__ a, const float* __restrict__ b,
                                  unsigned short* __restrict__ c, int n) {
  int i = blockIdx.x * blockDim.x + threadIdx.x;
  if (i < n) c[i] = f2bf(a[i] + b[i]);
}

// h[b,t,:] += pos[t,:] + timeVector[b,:]
__global__ void add_pos_time_kernel(float* __restrict__ h, const float* __restrict__ pos,
                                    const float* __restrict__ tv) {
  int i = blockIdx.x * blockDim.x + threadIdx.x;
  if (i >= T_TOK * D_MODEL) return;
  int row = i >> 9, d = i & 511;
  int b = row >> 9, t = row & 511;
  h[i] += pos[t * D_MODEL + d] + tv[b * D_MODEL + d];
}

// h[b,t,:] += co[b,:]   (cross-attn with single KV token => broadcast V path)
__global__ void bcast_add_kernel(float* __restrict__ h, const float* __restrict__ co) {
  int i = blockIdx.x * blockDim.x + threadIdx.x;
  if (i >= T_TOK * D_MODEL) return;
  int b = (i >> 9) >> 9, d = i & 511;
  h[i] += co[b * D_MODEL + d];
}

// LayerNorm over D=512, bf16 output (register-resident row: 16 f32/lane)
__global__ __launch_bounds__(256) void layernorm_bf16_kernel(
    const float* __restrict__ x, const float* __restrict__ g,
    const float* __restrict__ bta, unsigned short* __restrict__ y, int rows) {
  int row = blockIdx.x * 8 + (threadIdx.x >> 5);
  int lane = threadIdx.x & 31;
  if (row >= rows) return;
  const float* xr = x + (long)row * D_MODEL;
  float buf[16], s = 0.f;
#pragma unroll
  for (int i = 0; i < 16; ++i) { buf[i] = xr[lane + 32 * i]; s += buf[i]; }
#pragma unroll
  for (int o = 16; o; o >>= 1) s += __shfl_xor(s, o, 32);
  float mean = s * (1.0f / D_MODEL);
  float v = 0.f;
#pragma unroll
  for (int i = 0; i < 16; ++i) { float d = buf[i] - mean; v += d * d; }
#pragma unroll
  for (int o = 16; o; o >>= 1) v += __shfl_xor(v, o, 32);
  float rstd = rsqrtf(v * (1.0f / D_MODEL) + 1e-5f);
  unsigned short* yr = y + (long)row * D_MODEL;
#pragma unroll
  for (int i = 0; i < 16; ++i) {
    int c = lane + 32 * i;
    yr[c] = f2bf((buf[i] - mean) * rstd * g[c] + bta[c]);
  }
}

// row softmax over 512, f32 in -> bf16 out (register-resident row)
__global__ __launch_bounds__(256) void softmax512_bf16_kernel(
    const float* __restrict__ s, unsigned short* __restrict__ o16, int rows) {
  int row = blockIdx.x * 8 + (threadIdx.x >> 5);
  int lane = threadIdx.x & 31;
  if (row >= rows) return;
  const float* r = s + (long)row * 512;
  float buf[16], mx = -1e30f;
#pragma unroll
  for (int i = 0; i < 16; ++i) { buf[i] = r[lane + 32 * i]; mx = fmaxf(mx, buf[i]); }
#pragma unroll
  for (int o = 16; o; o >>= 1) mx = fmaxf(mx, __shfl_xor(mx, o, 32));
  float sum = 0.f;
#pragma unroll
  for (int i = 0; i < 16; ++i) { buf[i] = expf(buf[i] - mx); sum += buf[i]; }
#pragma unroll
  for (int o = 16; o; o >>= 1) sum += __shfl_xor(sum, o, 32);
  float inv = 1.0f / sum;
  unsigned short* w = o16 + (long)row * 512;
#pragma unroll
  for (int i = 0; i < 16; ++i) w[lane + 32 * i] = f2bf(buf[i] * inv);
}

// logits[t,e] = nm[t,:]@wr[:,e] + br[e] + cond[b(t),:]@wrc[:,e]   (nm in bf16)
__global__ void moe_logits_kernel(const unsigned short* __restrict__ x,
                                  const float* __restrict__ cond,
                                  const float* __restrict__ wr, const float* __restrict__ br,
                                  const float* __restrict__ wrc, float* __restrict__ logits) {
  int idx = blockIdx.x * blockDim.x + threadIdx.x;
  if (idx >= T_TOK * N_EXP) return;
  int t = idx >> 3, e = idx & 7;
  const unsigned short* xr = x + (long)t * D_MODEL;
  const float* cr = cond + (long)(t >> 9) * D_MODEL;
  float s = br[e];
  for (int d = 0; d < D_MODEL; ++d)
    s += bf2f(xr[d]) * wr[d * N_EXP + e] + cr[d] * wrc[d * N_EXP + e];
  logits[idx] = s;
}

// softmax over E, top-2, normalized weights scattered into wfull[t,8]
__global__ void router_kernel(const float* __restrict__ logits, float* __restrict__ wfull) {
  int t = blockIdx.x * blockDim.x + threadIdx.x;
  if (t >= T_TOK) return;
  const float* l = logits + t * N_EXP;
  float p[N_EXP];
  float mx = l[0];
#pragma unroll
  for (int e = 1; e < N_EXP; ++e) mx = fmaxf(mx, l[e]);
  float se = 0.f;
#pragma unroll
  for (int e = 0; e < N_EXP; ++e) { p[e] = expf(l[e] - mx); se += p[e]; }
  float inv = 1.0f / se;
#pragma unroll
  for (int e = 0; e < N_EXP; ++e) p[e] *= inv;
  int i0 = 0;
#pragma unroll
  for (int e = 1; e < N_EXP; ++e) if (p[e] > p[i0]) i0 = e;
  int i1 = (i0 == 0) ? 1 : 0;
#pragma unroll
  for (int e = 0; e < N_EXP; ++e) if (e != i0 && p[e] > p[i1]) i1 = e;
  float denom = fmaxf(p[i0] + p[i1], 1e-6f);
  float* w = wfull + t * N_EXP;
#pragma unroll
  for (int e = 0; e < N_EXP; ++e) w[e] = 0.f;
  w[i0] = p[i0] / denom;
  w[i1] = p[i1] / denom;
}

// aux = mean((imp-u)^2) + mean((load-u)^2); deterministic wave-tree reductions
__global__ __launch_bounds__(256) void moe_aux_kernel(
    const float* __restrict__ logits, const float* __restrict__ wfull,
    float* __restrict__ auxout) {
  __shared__ float simp[N_EXP], sload[N_EXP];
  int wave = threadIdx.x >> 5, lane = threadIdx.x & 31;  // wave == expert id
  float accI = 0.f, accL = 0.f;
  for (int t = lane; t < T_TOK; t += 32) {
    const float* l = logits + t * N_EXP;
    float mx = l[0];
#pragma unroll
    for (int e = 1; e < N_EXP; ++e) mx = fmaxf(mx, l[e]);
    float se = 0.f, pe = 0.f;
#pragma unroll
    for (int e = 0; e < N_EXP; ++e) {
      float ex = expf(l[e] - mx);
      se += ex;
      if (e == wave) pe = ex;
    }
    accI += pe / se;
    accL += wfull[t * N_EXP + wave];
  }
#pragma unroll
  for (int o = 16; o; o >>= 1) {
    accI += __shfl_xor(accI, o, 32);
    accL += __shfl_xor(accL, o, 32);
  }
  if (lane == 0) { simp[wave] = accI; sload[wave] = accL; }
  __syncthreads();
  if (threadIdx.x == 0) {
    float is = 0.f, ls = 0.f;
    for (int e = 0; e < N_EXP; ++e) { is += simp[e] / T_TOK; ls += sload[e]; }
    is = fmaxf(is, 1e-6f); ls = fmaxf(ls, 1e-6f);
    const float u = 1.0f / N_EXP;
    float aI = 0.f, aL = 0.f;
    for (int e = 0; e < N_EXP; ++e) {
      float d1 = (simp[e] / T_TOK) / is - u; aI += d1 * d1;
      float d2 = sload[e] / ls - u;          aL += d2 * d2;
    }
    auxout[0] = (aI + aL) * (1.0f / N_EXP);
  }
}

__global__ void aux_mean_kernel(const float* __restrict__ auxb, float* __restrict__ out) {
  out[OUT_ELEMS] = 0.25f * (auxb[0] + auxb[1] + auxb[2] + auxb[3]);
}

// ---------------------------------------------------------------------------
// Host-side orchestration
// ---------------------------------------------------------------------------
static void launch_gemm(hipStream_t st, const unsigned short* A, const unsigned short* B,
                        void* C, const float* bias, const float* rs, int rss,
                        int M, int N, int K, int lda, int ldb, int ldc, float alpha,
                        int nbz, int headsZ, long sAb, long sAh, long sBb, long sBh,
                        long sCb, long sCh, bool gelu, bool accum, bool out16) {
  dim3 grid((N + 63) / 64, (M + 127) / 128, nbz);
  dim3 block(128);
#define GEMM_CALL(GE, AC, BI, RSC, O16)                                               \
  gemm_wmma<GE, AC, BI, RSC, O16><<<grid, block, 0, st>>>(                            \
      A, B, C, bias, rs, rss, M, N, K, lda, ldb, ldc, alpha, headsZ, sAb, sAh, sBb,   \
      sBh, sCb, sCh)
  if (gelu)               GEMM_CALL(1, 0, 1, 0, 1);  // expert up-proj -> bf16
  else if (accum && rs)   GEMM_CALL(0, 1, 1, 1, 0);  // expert down-proj -> +=f32
  else if (accum)         GEMM_CALL(0, 1, 1, 0, 0);  // attn out-proj -> +=f32
  else if (bias && out16) GEMM_CALL(0, 0, 1, 0, 1);  // qkv / cv -> bf16
  else if (bias)          GEMM_CALL(0, 0, 1, 0, 0);  // f32 out w/ bias
  else if (out16)         GEMM_CALL(0, 0, 0, 0, 1);  // S@V -> bf16
  else                    GEMM_CALL(0, 0, 0, 0, 0);  // scores -> f32 (alpha)
#undef GEMM_CALL
}

static void launch_cvt(hipStream_t st, const float* src, unsigned short* dst, int n) {
  int n4 = n / 4;
  cvt_bf16_kernel<<<(n4 + 255) / 256, 256, 0, st>>>(src, dst, n4);
}

struct WsBufs {
  // f32
  float *H, *S, *LG, *WF, *COND, *CO, *AUXB;
  // bf16 activations
  unsigned short *NM16, *Q16, *K16, *KT16, *V16, *O16, *S16, *HB16, *CTX16, *CV16, *ROT16, *H16;
  // bf16 weight staging
  unsigned short *BW_WQ, *BW_WK, *BW_WV, *BW_WO, *BW_W1, *BW_W2;          // per-block (reused)
  unsigned short *SW_WIN, *SW_WCOND, *SW_WOUT, *SW_CWV, *SW_CWO;          // static
};
static WsBufs carve(void* d_ws) {
  WsBufs w;
  float* p = (float*)d_ws;
  w.H    = p; p += (size_t)T_TOK * D_MODEL;
  w.S    = p; p += (size_t)NB * N_HEADS * SEQ * SEQ;
  w.LG   = p; p += (size_t)T_TOK * N_EXP;
  w.WF   = p; p += (size_t)T_TOK * N_EXP;
  w.COND = p; p += (size_t)NB * D_MODEL;
  w.CO   = p; p += (size_t)NB * D_MODEL;
  w.AUXB = p; p += 8;
  // align bf16 area to 32B
  unsigned short* q = (unsigned short*)(((uintptr_t)p + 31) & ~(uintptr_t)31);
  w.NM16  = q; q += (size_t)T_TOK * D_MODEL;
  w.Q16   = q; q += (size_t)T_TOK * D_MODEL;
  w.K16   = q; q += (size_t)T_TOK * D_MODEL;
  w.KT16  = q; q += (size_t)T_TOK * D_MODEL;
  w.V16   = q; q += (size_t)T_TOK * D_MODEL;
  w.O16   = q; q += (size_t)T_TOK * D_MODEL;
  w.S16   = q; q += (size_t)NB * N_HEADS * SEQ * SEQ;
  w.HB16  = q; q += (size_t)T_TOK * FF_DIM;
  w.CTX16 = q; q += (size_t)NB * TD_DIM;
  w.CV16  = q; q += (size_t)NB * D_MODEL;
  w.ROT16 = q; q += (size_t)T_TOK * IN_CH;
  w.H16   = q; q += (size_t)T_TOK * D_MODEL;
  w.BW_WQ = q; q += (size_t)D_MODEL * D_MODEL;
  w.BW_WK = q; q += (size_t)D_MODEL * D_MODEL;
  w.BW_WV = q; q += (size_t)D_MODEL * D_MODEL;
  w.BW_WO = q; q += (size_t)D_MODEL * D_MODEL;
  w.BW_W1 = q; q += (size_t)N_EXP * D_MODEL * FF_DIM;
  w.BW_W2 = q; q += (size_t)N_EXP * FF_DIM * D_MODEL;
  w.SW_WIN   = q; q += (size_t)IN_CH * D_MODEL;
  w.SW_WCOND = q; q += (size_t)TD_DIM * D_MODEL;
  w.SW_WOUT  = q; q += (size_t)D_MODEL * IN_CH;
  w.SW_CWV   = q; q += (size_t)TD_DIM * D_MODEL;
  w.SW_CWO   = q; q += (size_t)D_MODEL * D_MODEL;
  return w;
}

// param leaf indices (setup_inputs() dict insertion order, recursive)
enum {
  IN_ROT = 0, IN_TIME = 1, IN_TEXT = 2, IN_TAG = 3,
  P_W_IN = 4, P_B_IN = 5, P_POS = 6, P_W_COND = 7, P_B_COND = 8,
  P_DOWN0 = 9, P_DOWN1 = 28, P_UP0 = 47, P_UP1 = 66, P_CROSS = 85,
  P_W_OUT = 95, P_B_OUT = 96
};
enum {
  L_LN1G = 0, L_LN1B, L_WQ, L_BQ, L_WK, L_BK, L_WV, L_BV, L_WO, L_BO,
  L_LN2G, L_LN2B, L_WR, L_BR, L_WRC, L_W1, L_B1, L_W2, L_B2
};

static void run_block(hipStream_t st, void* const* din, int base, const WsBufs& w, int auxslot) {
  auto P = [&](int i) { return (const float*)din[base + i]; };
  const int TPB = 256;
  const float inv_sqrt_hd = 0.08838834764831845f;  // 1/sqrt(128)

  // stage this block's matmul weights as bf16
  launch_cvt(st, P(L_WQ), w.BW_WQ, D_MODEL * D_MODEL);
  launch_cvt(st, P(L_WK), w.BW_WK, D_MODEL * D_MODEL);
  launch_cvt(st, P(L_WV), w.BW_WV, D_MODEL * D_MODEL);
  launch_cvt(st, P(L_WO), w.BW_WO, D_MODEL * D_MODEL);
  launch_cvt(st, P(L_W1), w.BW_W1, N_EXP * D_MODEL * FF_DIM);
  launch_cvt(st, P(L_W2), w.BW_W2, N_EXP * FF_DIM * D_MODEL);

  // nm = LN1(h)
  layernorm_bf16_kernel<<<T_TOK / 8, TPB, 0, st>>>(w.H, P(L_LN1G), P(L_LN1B), w.NM16, T_TOK);
  // q,k,v = nm@W + b  (bf16 out)
  launch_gemm(st, w.NM16, w.BW_WQ, w.Q16, P(L_BQ), nullptr, 0, T_TOK, D_MODEL, D_MODEL,
              D_MODEL, D_MODEL, D_MODEL, 1.0f, 1, 1, 0, 0, 0, 0, 0, 0, false, false, true);
  launch_gemm(st, w.NM16, w.BW_WK, w.K16, P(L_BK), nullptr, 0, T_TOK, D_MODEL, D_MODEL,
              D_MODEL, D_MODEL, D_MODEL, 1.0f, 1, 1, 0, 0, 0, 0, 0, 0, false, false, true);
  launch_gemm(st, w.NM16, w.BW_WV, w.V16, P(L_BV), nullptr, 0, T_TOK, D_MODEL, D_MODEL,
              D_MODEL, D_MODEL, D_MODEL, 1.0f, 1, 1, 0, 0, 0, 0, 0, 0, false, false, true);
  // KT[b][d][t] = K[b][t][d]  -> score GEMM becomes NN
  {
    dim3 tg(16, 16, NB), tb(32, 8);
    transpose512_kernel<<<tg, tb, 0, st>>>(w.K16, w.KT16);
  }
  // S[b,h] = (Q_bh @ KT_bh) * scale  (32 batched NN GEMMs, f32 out)
  launch_gemm(st, w.Q16, w.KT16, w.S, nullptr, nullptr, 0, SEQ, SEQ, HEAD_D,
              D_MODEL, SEQ, SEQ, inv_sqrt_hd, NB * N_HEADS, N_HEADS,
              (long)SEQ * D_MODEL, HEAD_D,                      // A: Q slice
              (long)SEQ * D_MODEL, (long)HEAD_D * SEQ,          // B: KT slice
              (long)N_HEADS * SEQ * SEQ, (long)SEQ * SEQ,       // C: S slice
              false, false, false);
  softmax512_bf16_kernel<<<(NB * N_HEADS * SEQ) / 8, TPB, 0, st>>>(
      w.S, w.S16, NB * N_HEADS * SEQ);
  // O[b,h] = S @ V_bh  (bf16 out)
  launch_gemm(st, w.S16, w.V16, w.O16, nullptr, nullptr, 0, SEQ, HEAD_D, SEQ,
              SEQ, D_MODEL, D_MODEL, 1.0f, NB * N_HEADS, N_HEADS,
              (long)N_HEADS * SEQ * SEQ, (long)SEQ * SEQ,
              (long)SEQ * D_MODEL, HEAD_D, (long)SEQ * D_MODEL, HEAD_D, false, false, true);
  // h += O @ Wo + bo
  launch_gemm(st, w.O16, w.BW_WO, w.H, P(L_BO), nullptr, 0, T_TOK, D_MODEL, D_MODEL,
              D_MODEL, D_MODEL, D_MODEL, 1.0f, 1, 1, 0, 0, 0, 0, 0, 0, false, true, false);
  // nm = LN2(h)
  layernorm_bf16_kernel<<<T_TOK / 8, TPB, 0, st>>>(w.H, P(L_LN2G), P(L_LN2B), w.NM16, T_TOK);
  // MoE router + aux
  moe_logits_kernel<<<(T_TOK * N_EXP + TPB - 1) / TPB, TPB, 0, st>>>(
      w.NM16, w.COND, P(L_WR), P(L_BR), P(L_WRC), w.LG);
  router_kernel<<<(T_TOK + TPB - 1) / TPB, TPB, 0, st>>>(w.LG, w.WF);
  moe_aux_kernel<<<1, TPB, 0, st>>>(w.LG, w.WF, w.AUXB + auxslot);
  // dense experts: h += wfull[:,e] * (gelu(nm@W1e+b1e)@W2e + b2e)
  for (int e = 0; e < N_EXP; ++e) {
    const unsigned short* w1 = w.BW_W1 + (size_t)e * D_MODEL * FF_DIM;
    const unsigned short* w2 = w.BW_W2 + (size_t)e * FF_DIM * D_MODEL;
    const float* b1 = P(L_B1) + (size_t)e * FF_DIM;
    const float* b2 = P(L_B2) + (size_t)e * D_MODEL;
    launch_gemm(st, w.NM16, w1, w.HB16, b1, nullptr, 0, T_TOK, FF_DIM, D_MODEL,
                D_MODEL, FF_DIM, FF_DIM, 1.0f, 1, 1, 0, 0, 0, 0, 0, 0, true, false, true);
    launch_gemm(st, w.HB16, w2, w.H, b2, w.WF + e, N_EXP, T_TOK, D_MODEL, FF_DIM,
                FF_DIM, D_MODEL, D_MODEL, 1.0f, 1, 1, 0, 0, 0, 0, 0, 0, false, true, false);
  }
}

extern "C" void kernel_launch(void* const* d_in, const int* in_sizes, int n_in,
                              void* d_out, int out_size, void* d_ws, size_t ws_size,
                              hipStream_t stream) {
  (void)in_sizes; (void)n_in; (void)out_size; (void)ws_size;
  WsBufs w = carve(d_ws);
  const int TPB = 256;
  const int NEL = T_TOK * D_MODEL;

  // static weights -> bf16
  launch_cvt(stream, (const float*)d_in[P_W_IN], w.SW_WIN, IN_CH * D_MODEL);
  launch_cvt(stream, (const float*)d_in[P_W_COND], w.SW_WCOND, TD_DIM * D_MODEL);
  launch_cvt(stream, (const float*)d_in[P_W_OUT], w.SW_WOUT, D_MODEL * IN_CH);
  launch_cvt(stream, (const float*)d_in[P_CROSS + 6], w.SW_CWV, TD_DIM * D_MODEL);
  launch_cvt(stream, (const float*)d_in[P_CROSS + 8], w.SW_CWO, D_MODEL * D_MODEL);
  launch_cvt(stream, (const float*)d_in[IN_ROT], w.ROT16, T_TOK * IN_CH);

  // ctx = text + tag (bf16) ; cond = ctx @ w_cond + b_cond (f32)
  ewadd_bf16_kernel<<<(NB * TD_DIM + TPB - 1) / TPB, TPB, 0, stream>>>(
      (const float*)d_in[IN_TEXT], (const float*)d_in[IN_TAG], w.CTX16, NB * TD_DIM);
  launch_gemm(stream, w.CTX16, w.SW_WCOND, w.COND, (const float*)d_in[P_B_COND],
              nullptr, 0, NB, D_MODEL, TD_DIM, TD_DIM, D_MODEL, D_MODEL, 1.0f,
              1, 1, 0, 0, 0, 0, 0, 0, false, false, false);
  // h = rot6d @ w_in + b_in + pos + time
  launch_gemm(stream, w.ROT16, w.SW_WIN, w.H, (const float*)d_in[P_B_IN],
              nullptr, 0, T_TOK, D_MODEL, IN_CH, IN_CH, D_MODEL, D_MODEL, 1.0f,
              1, 1, 0, 0, 0, 0, 0, 0, false, false, false);
  add_pos_time_kernel<<<(NEL + TPB - 1) / TPB, TPB, 0, stream>>>(
      w.H, (const float*)d_in[P_POS], (const float*)d_in[IN_TIME]);

  // down blocks
  run_block(stream, d_in, P_DOWN0, w, 0);
  run_block(stream, d_in, P_DOWN1, w, 1);

  // cross-attention: single KV token per batch -> softmax == 1 -> pure V path
  launch_gemm(stream, w.CTX16, w.SW_CWV, w.CV16, (const float*)d_in[P_CROSS + 7],
              nullptr, 0, NB, D_MODEL, TD_DIM, TD_DIM, D_MODEL, D_MODEL, 1.0f,
              1, 1, 0, 0, 0, 0, 0, 0, false, false, true);
  launch_gemm(stream, w.CV16, w.SW_CWO, w.CO, (const float*)d_in[P_CROSS + 9],
              nullptr, 0, NB, D_MODEL, D_MODEL, D_MODEL, D_MODEL, D_MODEL, 1.0f,
              1, 1, 0, 0, 0, 0, 0, 0, false, false, false);
  bcast_add_kernel<<<(NEL + TPB - 1) / TPB, TPB, 0, stream>>>(w.H, w.CO);

  // up blocks
  run_block(stream, d_in, P_UP0, w, 2);
  run_block(stream, d_in, P_UP1, w, 3);

  // out = h @ w_out + b_out  -> d_out ; aux scalar appended
  launch_cvt(stream, w.H, w.H16, T_TOK * D_MODEL);
  launch_gemm(stream, w.H16, w.SW_WOUT, (float*)d_out, (const float*)d_in[P_B_OUT],
              nullptr, 0, T_TOK, IN_CH, D_MODEL, D_MODEL, IN_CH, IN_CH, 1.0f,
              1, 1, 0, 0, 0, 0, 0, 0, false, false, false);
  aux_mean_kernel<<<1, 1, 0, stream>>>(w.AUXB, (float*)d_out);
}
